// _MatrixDecompositionBase_4690104287363
// MI455X (gfx1250) — compile-verified, matrix operand order checked by reading, not run on hardware
//
#include <hip/hip_runtime.h>

typedef __attribute__((ext_vector_type(16))) __bf16 v16bf;
typedef __attribute__((ext_vector_type(8)))  __bf16 v8bf;
typedef __attribute__((ext_vector_type(8)))  float  v8f;

#define EPS 1e-6f

// ---------------------------------------------------------------------------
// Elementwise helpers
// ---------------------------------------------------------------------------
__global__ void k_f32_to_bf16(const float* __restrict__ x, __bf16* __restrict__ y, long n) {
    long i = (long)blockIdx.x * blockDim.x + threadIdx.x;
    long stride = (long)gridDim.x * blockDim.x;
    for (; i < n; i += stride) y[i] = (__bf16)x[i];
}

// bases: (B, D, R) fp32.  Normalize along D: b / (||b||_2 + EPS). Write fp32 + bf16.
__global__ void k_norm_bases(const float* __restrict__ bases,
                             float* __restrict__ bF, __bf16* __restrict__ bH,
                             int D, int R) {
    int b = blockIdx.x / R;
    int r = blockIdx.x % R;
    const float* src = bases + (long)b * D * R + r;
    __shared__ float red[64];
    float ss = 0.f;
    for (int d = threadIdx.x; d < D; d += 64) {
        float v = src[(long)d * R];
        ss += v * v;
    }
    red[threadIdx.x] = ss;
    __syncthreads();
    for (int s = 32; s > 0; s >>= 1) {
        if ((int)threadIdx.x < s) red[threadIdx.x] += red[threadIdx.x + s];
        __syncthreads();
    }
    float inv = 1.0f / (sqrtf(red[0]) + EPS);
    for (int d = threadIdx.x; d < D; d += 64) {
        long idx = (long)b * D * R + (long)d * R + r;
        float v = src[(long)d * R] * inv;
        bF[idx] = v;
        bH[idx] = (__bf16)v;
    }
}

// rowwise softmax over R contiguous elements of invt*num; writes fp32 + bf16.
__global__ void k_softmax_rows(const float* __restrict__ num,
                               float* __restrict__ cF, __bf16* __restrict__ cH,
                               long rows, int R, float invt) {
    long row = (long)blockIdx.x * blockDim.x + threadIdx.x;
    if (row >= rows) return;
    const float* p = num + row * R;
    float mx = -3.0e38f;
    for (int r = 0; r < R; ++r) mx = fmaxf(mx, invt * p[r]);
    float s = 0.f;
    for (int r = 0; r < R; ++r) s += __expf(invt * p[r] - mx);
    float inv = 1.0f / s;
    for (int r = 0; r < R; ++r) {
        float v = __expf(invt * p[r] - mx) * inv;
        cF[row * R + r] = v;
        cH[row * R + r] = (__bf16)v;
    }
}

// z = z * num / (den + eps); keep fp32 master, refresh bf16 shadow.
__global__ void k_mul_update(float* __restrict__ zF, __bf16* __restrict__ zH,
                             const float* __restrict__ num, const float* __restrict__ den,
                             long n) {
    long i = (long)blockIdx.x * blockDim.x + threadIdx.x;
    long stride = (long)gridDim.x * blockDim.x;
    for (; i < n; i += stride) {
        float v = zF[i] * num[i] / (den[i] + EPS);
        zF[i] = v;
        zH[i] = (__bf16)v;
    }
}

// ---------------------------------------------------------------------------
// WMMA bf16 GEMM, 64x64 tile per 128-thread block (4 waves), KC=32.
// MODE 0 (TN): C[m,n] = sum_k A[k*lda+m] * B[k*ldb+n]
// MODE 1 (NN): C[m,n] = sum_k A[m*lda+k] * B[k*ldb+n]
// MODE 2 (NT): C[m,n] = sum_k A[m*lda+k] * B[n*ldb+k]
// Requires M%64==0, N%64==0, K%32==0 (true for all call sites here).
// ---------------------------------------------------------------------------
template <int MODE>
__global__ __launch_bounds__(128) void k_wmma_gemm(
    const __bf16* __restrict__ A, int lda, long batchStrideA,
    const __bf16* __restrict__ B, int ldb, long batchStrideB,
    float* __restrict__ Cf, __bf16* __restrict__ Ch, int ldc, long batchStrideC,
    int K) {
    constexpr int KC = 32;
    constexpr int LSTR = 40;  // 32 + 8 pad (bf16 elems); row pitch 80B (16B aligned)

    __shared__ __attribute__((aligned(16))) __bf16 shA[64 * LSTR];
    __shared__ __attribute__((aligned(16))) __bf16 shB[64 * LSTR];

    const int bz = blockIdx.z;
    const int m0 = blockIdx.x * 64;
    const int n0 = blockIdx.y * 64;
    A += (long)bz * batchStrideA;
    B += (long)bz * batchStrideB;

    const int t    = threadIdx.x;   // 0..127
    const int lane = t & 31;
    const int wave = t >> 5;        // 0..3 -> 16-row strip
    const int g    = lane >> 4;     // lane group (K-half select)
    const int mr   = lane & 15;     // row (A) / col (B) within 16

    v8f acc[4];
#pragma unroll
    for (int j = 0; j < 4; ++j)
#pragma unroll
        for (int i = 0; i < 8; ++i) acc[j][i] = 0.f;

    for (int k0 = 0; k0 < K; k0 += KC) {
        // ---- stage A tile into shA[m][k] ----
        if (MODE == 0) {  // A[k*lda + m]: transpose on store
            int kr = t >> 2;
            int ms = (t & 3) * 16;
            const __bf16* src = A + (long)(k0 + kr) * lda + (m0 + ms);
#pragma unroll
            for (int i = 0; i < 16; ++i) shA[(ms + i) * LSTR + kr] = src[i];
        } else {          // A[m*lda + k]: direct
            int mrw = t >> 1;
            int ks  = (t & 1) * 16;
            const __bf16* src = A + (long)(m0 + mrw) * lda + (k0 + ks);
#pragma unroll
            for (int i = 0; i < 16; ++i) shA[mrw * LSTR + ks + i] = src[i];
        }
        // ---- stage B tile into shB[n][k] ----
        if (MODE == 2) {  // B[n*ldb + k]: direct
            int nrw = t >> 1;
            int ks  = (t & 1) * 16;
            const __bf16* src = B + (long)(n0 + nrw) * ldb + (k0 + ks);
#pragma unroll
            for (int i = 0; i < 16; ++i) shB[nrw * LSTR + ks + i] = src[i];
        } else {          // B[k*ldb + n]: transpose on store
            int kr = t >> 2;
            int ns = (t & 3) * 16;
            const __bf16* src = B + (long)(k0 + kr) * ldb + (n0 + ns);
#pragma unroll
            for (int i = 0; i < 16; ++i) shB[(ns + i) * LSTR + kr] = src[i];
        }
        __syncthreads();

        // ---- A fragment: row = wave*16 + mr; lane group g takes K {8g..8g+7, 16+8g..16+8g+7}
        v16bf af;
        {
            const int arow = wave * 16 + mr;
            v8bf lo = *(const v8bf*)&shA[arow * LSTR + 8 * g];
            v8bf hi = *(const v8bf*)&shA[arow * LSTR + 16 + 8 * g];
#pragma unroll
            for (int i = 0; i < 8; ++i) { af[i] = lo[i]; af[8 + i] = hi[i]; }
        }
#pragma unroll
        for (int j = 0; j < 4; ++j) {
            const int brow = j * 16 + mr;
            v16bf bf;
            v8bf lo = *(const v8bf*)&shB[brow * LSTR + 8 * g];
            v8bf hi = *(const v8bf*)&shB[brow * LSTR + 16 + 8 * g];
#pragma unroll
            for (int i = 0; i < 8; ++i) { bf[i] = lo[i]; bf[8 + i] = hi[i]; }
            acc[j] = __builtin_amdgcn_wmma_f32_16x16x32_bf16(
                false, af, false, bf, (short)0, acc[j], false, false);
        }
        __syncthreads();
    }

    // ---- epilogue: VGPR i holds M = i + 8*g; lane mr holds N-col ----
    const long cbase = (long)bz * batchStrideC;
#pragma unroll
    for (int j = 0; j < 4; ++j)
#pragma unroll
        for (int i = 0; i < 8; ++i) {
            int row = m0 + wave * 16 + i + 8 * g;
            int col = n0 + j * 16 + mr;
            float v = acc[j][i];
            if (Cf) Cf[cbase + (long)row * ldc + col] = v;
            if (Ch) Ch[cbase + (long)row * ldc + col] = (__bf16)v;
        }
}

// ---------------------------------------------------------------------------
// Host orchestration
// ---------------------------------------------------------------------------
extern "C" void kernel_launch(void* const* d_in, const int* in_sizes, int n_in,
                              void* d_out, int out_size, void* d_ws, size_t ws_size,
                              hipStream_t stream) {
    const float* x        = (const float*)d_in[0];  // (32, 512, 64, 64)
    const float* bases_in = (const float*)d_in[1];  // (32, 512, 64)

    const int B = 32, D = 512, R = 64, N = 4096;
    const int STEPS = 6;

    // ---- carve workspace (~267 MB) ----
    char* p = (char*)d_ws;
    auto take = [&](size_t bytes) -> void* {
        void* q = (void*)p;
        p += (bytes + 255) & ~(size_t)255;
        return q;
    };
    __bf16* xb  = (__bf16*)take((size_t)B * D * N * 2);  // bf16 X cache (128MB, fits 192MB L2)
    float*  bF  = (float*) take((size_t)B * D * R * 4);  // bases fp32 master
    __bf16* bH  = (__bf16*)take((size_t)B * D * R * 2);  // bases bf16 shadow
    float*  cF  = (float*) take((size_t)B * N * R * 4);  // coef fp32 master
    __bf16* cH  = (__bf16*)take((size_t)B * N * R * 2);  // coef bf16 shadow
    float*  numN = (float*)take((size_t)B * N * R * 4);
    float*  denN = (float*)take((size_t)B * N * R * 4);
    float*  numD = (float*)take((size_t)B * D * R * 4);
    float*  denD = (float*)take((size_t)B * D * R * 4);
    __bf16* Gh   = (__bf16*)take((size_t)B * R * R * 2); // bases^T bases (bf16)
    __bf16* C2h  = (__bf16*)take((size_t)B * R * R * 2); // coef^T coef  (bf16)

    const long sX = (long)D * N;   // 2M elems / batch
    const long sB = (long)D * R;
    const long sC = (long)N * R;
    const long sG = (long)R * R;

    dim3 blk(128);

    // 1) cast X to bf16 once
    k_f32_to_bf16<<<16384, 256, 0, stream>>>(x, xb, (long)B * D * N);
    // 2) normalize bases along D
    k_norm_bases<<<B * R, 64, 0, stream>>>(bases_in, bF, bH, D, R);

    // 3) init coef = softmax(INV_T * X^T bases)   [TN: M=4096, Nc=64, K=512]
    k_wmma_gemm<0><<<dim3(N / 64, 1, B), blk, 0, stream>>>(
        xb, N, sX, bH, R, sB, numN, nullptr, R, sC, D);
    k_softmax_rows<<<(unsigned)(((long)B * N + 255) / 256), 256, 0, stream>>>(
        numN, cF, cH, (long)B * N, R, 100.0f);

    auto coef_update = [&]() {
        // Gh = bases^T bases   [TN: M=64, Nc=64, K=512]
        k_wmma_gemm<0><<<dim3(1, 1, B), blk, 0, stream>>>(
            bH, R, sB, bH, R, sB, nullptr, Gh, R, sG, D);
        // numN = X^T bases     [TN: M=4096, Nc=64, K=512]
        k_wmma_gemm<0><<<dim3(N / 64, 1, B), blk, 0, stream>>>(
            xb, N, sX, bH, R, sB, numN, nullptr, R, sC, D);
        // denN = coef @ Gh     [NN: M=4096, Nc=64, K=64]
        k_wmma_gemm<1><<<dim3(N / 64, 1, B), blk, 0, stream>>>(
            cH, R, sC, Gh, R, sG, denN, nullptr, R, sC, R);
        // coef *= numN / (denN + eps)
        k_mul_update<<<8192, 256, 0, stream>>>(cF, cH, numN, denN, (long)B * N * R);
    };

    for (int it = 0; it < STEPS; ++it) {
        coef_update();
        // C2h = coef^T coef    [TN: M=64, Nc=64, K=4096]
        k_wmma_gemm<0><<<dim3(1, 1, B), blk, 0, stream>>>(
            cH, R, sC, cH, R, sC, nullptr, C2h, R, sG, N);
        // numD = X @ coef      [NN: M=512, Nc=64, K=4096]
        k_wmma_gemm<1><<<dim3(D / 64, 1, B), blk, 0, stream>>>(
            xb, N, sX, cH, R, sC, numD, nullptr, R, sB, N);
        // denD = bases @ C2h   [NN: M=512, Nc=64, K=64]
        k_wmma_gemm<1><<<dim3(D / 64, 1, B), blk, 0, stream>>>(
            bH, R, sB, C2h, R, sG, denD, nullptr, R, sB, R);
        // bases *= numD / (denD + eps)
        k_mul_update<<<2048, 256, 0, stream>>>(bF, bH, numD, denD, (long)B * D * R);
    }

    // final compute_coef (grad-path update in reference)
    coef_update();

    // reconstruction: out = bases @ coef^T  [NT: M=512, Nc=4096, K=64] -> fp32 d_out
    k_wmma_gemm<2><<<dim3(D / 64, N / 64, B), blk, 0, stream>>>(
        bH, R, sB, cH, R, sC, (float*)d_out, nullptr, N, sX, R);
}
